// RGB_Depth_Agreement_Refined_33835752358434
// MI455X (gfx1250) — matrix operand order repaired and checked by reading, not compiled, not run on hardware
//
#include <hip/hip_runtime.h>
#include <hip/hip_bf16.h>

typedef __attribute__((ext_vector_type(16))) _Float16 v16h;
typedef __attribute__((ext_vector_type(8)))  float    v8f;

#define NH 12
#define DM 768
#define HD 64
#define NB 16
#define SQ 577
#define SP 608                    // padded sequence: 19*32, multiple of 32
#define BS (NB * SQ)              // 9232 rows (577 tiles of 16, exact)
#define D3 (3 * DM)               // 2304

// workspace layout (element offsets):
//   qkv_ws : _Float16, 3 section blocks of [2 mod][NB][NH][SP*HD]:
//            sect 0 (Q): [SP][HD] row-major
//            sect 1 (K): [SP][HD] row-major
//            sect 2 (V): [HD][SP] TRANSPOSED (so PV B-tiles are contiguous)
//   o_ws   : _Float16 [2 mod][BS][DM]
static constexpr size_t QKV_ELEMS = (size_t)3 * 2 * NB * NH * SP * HD;
static constexpr size_t O_ELEMS   = (size_t)2 * BS * DM;

// ---------------- WMMA helpers (contiguous k order in A/B, wave32) -------

__device__ __forceinline__ v8f wmma32(v16h a, v16h b, v8f c) {
  // D = A(16x32 f16) * B(32x16 f16) + C(16x16 f32)
  return __builtin_amdgcn_wmma_f32_16x16x32_f16(
      false, a, false, b, (short)0, c, false, false);
}

// A tile 16(M) x 32(K) from row-major f32, ld = row stride.
// lanes 0-15: M=lane, halves i -> K=i (0..15)
// lanes16-31: M=lane-16, halves i -> K=16+i
__device__ __forceinline__ v16h load_A_f32(const float* base, int ld) {
  int l = threadIdx.x & 31;
  const float* row = base + (size_t)(l & 15) * ld + ((l >> 4) << 4);
  v16h a;
#pragma unroll
  for (int i = 0; i < 8; ++i) a[i] = (_Float16)row[i];
#pragma unroll
  for (int i = 0; i < 8; ++i) a[8 + i] = (_Float16)row[8 + i];
  return a;
}

__device__ __forceinline__ v16h load_A_f16(const _Float16* base, int ld) {
  int l = threadIdx.x & 31;
  const _Float16* row = base + (size_t)(l & 15) * ld + ((l >> 4) << 4);
  v16h a;
#pragma unroll
  for (int i = 0; i < 8; ++i) a[i] = row[i];
#pragma unroll
  for (int i = 0; i < 8; ++i) a[8 + i] = row[8 + i];
  return a;
}

// B tile 32(K) x 16(N) where memory holds B^T row-major: elem = base[n*ld+k].
// lanes 0-15: N=lane, halves i -> K=i;  lanes 16-31: N=lane-16, K=16+i.
// (Covers y = x @ W^T weight tiles, K^T score tiles, and V^T PV tiles.)
__device__ __forceinline__ v16h load_Bt_f32(const float* base, int ld) {
  int l = threadIdx.x & 31;
  const float* row = base + (size_t)(l & 15) * ld + ((l >> 4) << 4);
  v16h b;
#pragma unroll
  for (int i = 0; i < 16; ++i) b[i] = (_Float16)row[i];
  return b;
}

__device__ __forceinline__ v16h load_Bt_f16(const _Float16* base, int ld) {
  int l = threadIdx.x & 31;
  const _Float16* row = base + (size_t)(l & 15) * ld + ((l >> 4) << 4);
  v16h b;
#pragma unroll
  for (int i = 0; i < 16; ++i) b[i] = row[i];
  return b;
}

// ---------------- kernels ---------------------------------------------------

__global__ void zero_u4(uint4* p, size_t n) {
  size_t i = (size_t)blockIdx.x * blockDim.x + threadIdx.x;
  size_t stride = (size_t)gridDim.x * blockDim.x;
  uint4 z{0u, 0u, 0u, 0u};
  for (; i < n; i += stride) p[i] = z;
}

// QKV projection: [9232,768] @ [768,2304]^T + bias -> f16 QKV, head-major,
// seq padded to SP; V stored transposed. grid(577, 36, 2), block 128.
__global__ void qkv_gemm(const float* __restrict__ img,
                         const float* __restrict__ dpt,
                         const float* __restrict__ w_qkv,
                         const float* __restrict__ b_qkv,
                         _Float16* __restrict__ qkv) {
  int wave = threadIdx.x >> 5, lane = threadIdx.x & 31;
  int mt = blockIdx.x;                 // 0..576
  int nt = blockIdx.y * 4 + wave;      // 0..143
  int mod = blockIdx.z;
  const float* X = mod ? dpt : img;
  const float* Xt = X + (size_t)mt * 16 * DM;
  const float* Wt = w_qkv + (size_t)nt * 16 * DM;

  v8f acc = {};
#pragma unroll 4
  for (int k0 = 0; k0 < DM; k0 += 32) {
    v16h a = load_A_f32(Xt + k0, DM);
    v16h b = load_Bt_f32(Wt + k0, DM);
    acc = wmma32(a, b, acc);
  }

  int nl = lane & 15, mb = (lane >> 4) * 8;
  int ng = nt * 16 + nl;               // 0..2303
  int sect = ng / DM;                  // 0=Q 1=K 2=V (wave-uniform: 16|DM)
  int col = ng - sect * DM;
  int h = col >> 6, hd = col & 63;     // head index, head-dim (tile within head)
  float bias = b_qkv[ng];
  size_t blk = (((size_t)(sect * 2 + mod) * NB) /*+ bb later*/) ;
#pragma unroll
  for (int r = 0; r < 8; ++r) {
    int mg = mt * 16 + mb + r;         // < 9232
    int bb = mg / SQ, s = mg - bb * SQ;
    size_t bhbase = ((((size_t)(sect * 2 + mod) * NB + bb) * NH + h) * (size_t)SP) * HD;
    size_t dst = (sect < 2)
                     ? bhbase + (size_t)s * HD + hd      // Q, K: [SP][HD]
                     : bhbase + (size_t)hd * SP + s;     // V  : [HD][SP]
    qkv[dst] = (_Float16)(acc[r] + bias);
  }
}

// Fused dual attention + agreement. grid(37, 12, 16), block 256 (8 waves),
// dynamic LDS = 2 * 16 * SP floats (76 KB, fits 320 KB WGP LDS).
__global__ void attn(const _Float16* __restrict__ qkv,
                     _Float16* __restrict__ o_ws,
                     const float* __restrict__ alpha_raw,
                     const float* __restrict__ beta_raw) {
  extern __shared__ float p_buf[];               // [2][16][SP]
  int qt = blockIdx.x, h = blockIdx.y, b = blockIdx.z;
  int wave = threadIdx.x >> 5, lane = threadIdx.x & 31;
  int q0 = qt * 16;
  const float scale = 0.125f;                    // 1/sqrt(64)

  // base of SP*HD f16 block for (sect, mod)
  auto base = [&](int sect, int mod) -> const _Float16* {
    return qkv + (((size_t)(sect * 2 + mod) * NB + b) * NH + h) * (size_t)SP * HD;
  };

  // ---- Phase A: S = Q K^T * scale -> LDS (keys 0..591 cover all 577) ----
  for (int kt = wave; kt < 37; kt += 8) {
#pragma unroll
    for (int mod = 0; mod < 2; ++mod) {
      const _Float16* Q = base(0, mod);
      const _Float16* K = base(1, mod);
      v8f acc = {};
#pragma unroll
      for (int k0 = 0; k0 < HD; k0 += 32) {
        v16h a = load_A_f16(Q + (size_t)q0 * HD + k0, HD);
        v16h bm = load_Bt_f16(K + (size_t)kt * 16 * HD + k0, HD);
        acc = wmma32(a, bm, acc);
      }
      int nl = lane & 15, mb = (lane >> 4) * 8;
      float* P = p_buf + (size_t)mod * 16 * SP;
#pragma unroll
      for (int r = 0; r < 8; ++r)
        P[(mb + r) * SP + kt * 16 + nl] = acc[r] * scale;
    }
  }
  __syncthreads();

  // ---- Phase B: softmax x2, agreement softmax, refine (16 thr / row) ----
  {
    int row = threadIdx.x >> 4, sub = threadIdx.x & 15;
    float* ri = p_buf + row * SP;
    float* rd = p_buf + 16 * SP + row * SP;
    float alpha = 1.f / (1.f + __expf(-alpha_raw[0]));
    float beta  = 1.f / (1.f + __expf(-beta_raw[0]));

    // softmax(img scores)
    float mx = -1e30f;
    for (int c = sub; c < SQ; c += 16) mx = fmaxf(mx, ri[c]);
#pragma unroll
    for (int o = 8; o; o >>= 1) mx = fmaxf(mx, __shfl_xor(mx, o, 16));
    float sm = 0.f;
    for (int c = sub; c < SQ; c += 16) { float e = __expf(ri[c] - mx); ri[c] = e; sm += e; }
#pragma unroll
    for (int o = 8; o; o >>= 1) sm += __shfl_xor(sm, o, 16);
    float inv = 1.f / sm;
    for (int c = sub; c < SQ; c += 16) ri[c] *= inv;

    // softmax(dpt scores)
    mx = -1e30f;
    for (int c = sub; c < SQ; c += 16) mx = fmaxf(mx, rd[c]);
#pragma unroll
    for (int o = 8; o; o >>= 1) mx = fmaxf(mx, __shfl_xor(mx, o, 16));
    sm = 0.f;
    for (int c = sub; c < SQ; c += 16) { float e = __expf(rd[c] - mx); rd[c] = e; sm += e; }
#pragma unroll
    for (int o = 8; o; o >>= 1) sm += __shfl_xor(sm, o, 16);
    inv = 1.f / sm;
    for (int c = sub; c < SQ; c += 16) rd[c] *= inv;

    // agreement softmax + refinement (ar = sigmoid * p_ag + p)
    float mg = -1e30f;
    for (int c = sub; c < SQ; c += 16) mg = fmaxf(mg, ri[c] * rd[c]);
#pragma unroll
    for (int o = 8; o; o >>= 1) mg = fmaxf(mg, __shfl_xor(mg, o, 16));
    float sg = 0.f;
    for (int c = sub; c < SQ; c += 16) sg += __expf(ri[c] * rd[c] - mg);
#pragma unroll
    for (int o = 8; o; o >>= 1) sg += __shfl_xor(sg, o, 16);
    float invg = 1.f / sg;
    for (int c = sub; c < SQ; c += 16) {
      float pi = ri[c], pd = rd[c];
      float pag = __expf(pi * pd - mg) * invg;
      ri[c] = alpha * pag + pi;
      rd[c] = beta * pag + pd;
    }
    // zero padded key columns 577..607 so PV reads zeros
    for (int c = SQ + sub; c < SP; c += 16) { ri[c] = 0.f; rd[c] = 0.f; }
  }
  __syncthreads();

  // ---- Phase C: O = AR @ V. A: f32->f16 from LDS; B: contiguous rows of
  //      V^T ([HD][SP]) -> b128 loads. wave -> (mod, n-tile of 16 dims). ----
  {
    int mod = wave >> 2, nt = wave & 3;
    const float* P = p_buf + (size_t)mod * 16 * SP;
    const _Float16* Vt = base(2, mod);             // [HD][SP]
    v8f acc = {};
#pragma unroll 2
    for (int k0 = 0; k0 < SP; k0 += 32) {
      v16h a = load_A_f32(P + k0, SP);             // from LDS
      v16h bm = load_Bt_f16(Vt + (size_t)(nt * 16) * SP + k0, SP);
      acc = wmma32(a, bm, acc);
    }
    int nl = lane & 15, mb = (lane >> 4) * 8;
    int d = nt * 16 + nl;
#pragma unroll
    for (int r = 0; r < 8; ++r) {
      int s = q0 + mb + r;
      if (s < SQ) {
        size_t dst = ((size_t)mod * BS + (size_t)b * SQ + s) * DM + h * HD + d;
        o_ws[dst] = (_Float16)acc[r];
      }
    }
  }
}

// Output projection: O(f16) @ W_out^T + b -> fp32 d_out.
// grid(577, 12, 2), block 128 (4 waves; wave = N-tile).
__global__ void out_gemm(const _Float16* __restrict__ o_ws,
                         const float* __restrict__ w_out,
                         const float* __restrict__ b_out,
                         float* __restrict__ out) {
  int wave = threadIdx.x >> 5, lane = threadIdx.x & 31;
  int mt = blockIdx.x;                 // 0..576
  int nt = blockIdx.y * 4 + wave;      // 0..47
  int mod = blockIdx.z;
  const _Float16* A = o_ws + (size_t)mod * BS * DM + (size_t)mt * 16 * DM;
  const float* Wt = w_out + (size_t)nt * 16 * DM;

  v8f acc = {};
#pragma unroll 4
  for (int k0 = 0; k0 < DM; k0 += 32) {
    v16h a = load_A_f16(A + k0, DM);
    v16h b = load_Bt_f32(Wt + k0, DM);
    acc = wmma32(a, b, acc);
  }

  int nl = lane & 15, mb = (lane >> 4) * 8;
  int ng = nt * 16 + nl;
  float bias = b_out[ng];
#pragma unroll
  for (int r = 0; r < 8; ++r) {
    size_t mg = (size_t)mt * 16 + mb + r;
    out[(size_t)mod * BS * DM + mg * DM + ng] = acc[r] + bias;
  }
}

// ---------------- launch ----------------------------------------------------

extern "C" void kernel_launch(void* const* d_in, const int* in_sizes, int n_in,
                              void* d_out, int out_size, void* d_ws, size_t ws_size,
                              hipStream_t stream) {
  const float* img   = (const float*)d_in[0];
  const float* dpt   = (const float*)d_in[1];
  const float* w_qkv = (const float*)d_in[2];
  const float* b_qkv = (const float*)d_in[3];
  const float* w_out = (const float*)d_in[4];
  const float* b_out = (const float*)d_in[5];
  const float* a_raw = (const float*)d_in[6];
  const float* b_raw = (const float*)d_in[7];
  float* out = (float*)d_out;

  _Float16* qkv = (_Float16*)d_ws;
  _Float16* o_ws = qkv + QKV_ELEMS;

  // allow >64KB dynamic LDS for the attention kernel
  static_assert(2 * 16 * SP * sizeof(float) == 77824, "lds size");
  hipFuncSetAttribute((const void*)attn,
                      hipFuncAttributeMaxDynamicSharedMemorySize, 77824);

  // zero the QKV workspace (pad rows/cols must be 0 for score/PV tiles)
  size_t n4 = QKV_ELEMS * sizeof(_Float16) / sizeof(uint4);
  zero_u4<<<4096, 256, 0, stream>>>((uint4*)qkv, n4);

  qkv_gemm<<<dim3(577, 36, 2), 128, 0, stream>>>(img, dpt, w_qkv, b_qkv, qkv);

  attn<<<dim3(37, NH, NB), 256, 77824, stream>>>(qkv, o_ws, a_raw, b_raw);

  out_gemm<<<dim3(577, 12, 2), 128, 0, stream>>>(o_ws, w_out, b_out, out);
}
